// HierarchicalWorldModel_11836929867876
// MI455X (gfx1250) — compile-verified, matrix operand order checked by reading, not run on hardware
//
#include <hip/hip_runtime.h>

// ---------------- problem constants ----------------
#define B_  8
#define T_  256
#define S_  1024          // 4*T
#define D_  768
#define H_  12
#define DH_ 64
#define L_  6
#define DFF_ 3072
#define NC_ 256
#define QKVD_ (3*D_)      // 2304
#define EPS_ 1e-5f

typedef _Float16 h16 __attribute__((ext_vector_type(16)));
typedef _Float16 h8  __attribute__((ext_vector_type(8)));
typedef float    f8  __attribute__((ext_vector_type(8)));

// ---------------- WMMA helpers ----------------
// D = A(16x32 f16) x B(32x16 f16) + C(16x16 f32)
__device__ __forceinline__ f8 wmma16x16x32(h16 a, h16 b, f8 c) {
  return __builtin_amdgcn_wmma_f32_16x16x32_f16(false, a, false, b, (short)0, c, false, false);
}

// Load a 16x32 fragment from a K-major (row-contiguous-K) matrix tile.
// ISA layout (16-bit A / symmetric B): lane&15 = row; low half-wave holds
// K = kb..kb+7 and kb+16..kb+23 with kb = k (+8 for lanes 16..31).
__device__ __forceinline__ h16 frag_kmajor(const _Float16* t, long ld, int lane, int k) {
  int r  = lane & 15;
  int kb = k + ((lane & 16) ? 8 : 0);
  const h8* p0 = (const h8*)(t + (long)r * ld + kb);
  const h8* p1 = (const h8*)(t + (long)r * ld + kb + 16);
  h8 a = *p0, b = *p1;
  h16 out;
#pragma unroll
  for (int i = 0; i < 8; ++i) { out[i] = a[i]; out[i + 8] = b[i]; }
  return out;
}

// ---------------- fp32 -> fp16 convert ----------------
__global__ void cvt_f32_f16(const float* __restrict__ in, _Float16* __restrict__ out, long n) {
  long i = (long)blockIdx.x * blockDim.x + threadIdx.x;
  long st = (long)gridDim.x * blockDim.x;
  for (; i < n; i += st) out[i] = (_Float16)in[i];
}

// ---------------- embedding: interleaved [L0,L1,L2,A] + lvl + pos ----------------
__global__ void embed_kernel(const int* __restrict__ tokens, const int* __restrict__ actions,
                             const float* __restrict__ tok_emb, const float* __restrict__ act_emb,
                             const float* __restrict__ lvl_emb, const float* __restrict__ pos_emb,
                             float* __restrict__ x) {
  int bs = blockIdx.x;            // 0 .. B*S-1
  int b = bs / S_, s = bs % S_;
  int t = s >> 2, p = s & 3;
  const float* e;
  if (p < 3) e = tok_emb + ((size_t)p * NC_ + tokens[((size_t)b * T_ + t) * 3 + p]) * D_;
  else       e = act_emb + (size_t)actions[(size_t)b * T_ + t] * D_;
  float* xr = x + (size_t)bs * D_;
  const float* lv = lvl_emb + (size_t)p * D_;
  const float* ps = pos_emb + (size_t)s * D_;
#pragma unroll
  for (int j = 0; j < 3; ++j) {
    int d = threadIdx.x + j * 256;
    xr[d] = e[d] + lv[d] + ps[d];
  }
}

// ---------------- LayerNorm (exact fp32 stats) ----------------
__global__ void ln_kernel(const float* __restrict__ x, const float* __restrict__ g,
                          const float* __restrict__ bt, _Float16* __restrict__ out_h,
                          float* __restrict__ out_f, int write_f32) {
  int row = blockIdx.x, tid = threadIdx.x;
  const float* xr = x + (size_t)row * D_;
  float v0 = xr[tid], v1 = xr[tid + 256], v2 = xr[tid + 512];
  __shared__ float s1[256], s2[256];
  s1[tid] = v0 + v1 + v2;
  s2[tid] = v0 * v0 + v1 * v1 + v2 * v2;
  __syncthreads();
  for (int off = 128; off > 0; off >>= 1) {
    if (tid < off) { s1[tid] += s1[tid + off]; s2[tid] += s2[tid + off]; }
    __syncthreads();
  }
  float mean = s1[0] * (1.f / (float)D_);
  float var  = s2[0] * (1.f / (float)D_) - mean * mean;
  float rstd = rsqrtf(var + EPS_);
  float vv[3] = {v0, v1, v2};
#pragma unroll
  for (int j = 0; j < 3; ++j) {
    int d = tid + j * 256;
    float y = (vv[j] - mean) * rstd * g[d] + bt[d];
    if (write_f32) out_f[(size_t)row * D_ + d] = y;
    else           out_h[(size_t)row * D_ + d] = (_Float16)y;
  }
}

// ---------------- generic GEMM: C[M,N] = A[M,K] @ W[N,K]^T + bias ----------------
// Register-blocked 2x2: each wave owns a 32Mx32N block (4 accumulators,
// 2 A frags x 2 B frags per K step -> 4 WMMAs per fragment-load group).
// Block = 4 waves in a 2x2 grid -> 64x64 per workgroup.
// mode 0: fp16 out     (QKV)
// mode 1: GELU, fp16   (FFN1)
// mode 2: resid + fp32 (out-proj / FFN2), in-place on x allowed
__global__ void gemm_kernel(const _Float16* __restrict__ A, const _Float16* __restrict__ W,
                            const float* __restrict__ bias, _Float16* __restrict__ out_h,
                            float* __restrict__ out_f, const float* __restrict__ resid,
                            int M, int N, int K, int mode) {
  int lane = threadIdx.x & 31, wave = threadIdx.x >> 5;
  int wn = wave & 1, wm = wave >> 1;            // 2x2 wave grid within block
  int mt = blockIdx.y * 4 + wm * 2;             // two 16-row tiles: mt, mt+1
  int nt = blockIdx.x * 4 + wn * 2;             // two 16-col tiles: nt, nt+1
  const _Float16* A0 = A + (size_t)mt * 16 * K;
  const _Float16* A1 = A0 + (size_t)16 * K;
  const _Float16* W0 = W + (size_t)nt * 16 * K;
  const _Float16* W1 = W0 + (size_t)16 * K;
  f8 c00 = {0.f,0.f,0.f,0.f,0.f,0.f,0.f,0.f};
  f8 c01 = c00, c10 = c00, c11 = c00;
  for (int k = 0; k < K; k += 32) {
    __builtin_prefetch(A0 + (size_t)(lane & 15) * K + k + 64, 0, 1);
    __builtin_prefetch(W0 + (size_t)(lane & 15) * K + k + 64, 0, 1);
    h16 a0 = frag_kmajor(A0, K, lane, k);
    h16 a1 = frag_kmajor(A1, K, lane, k);
    h16 b0 = frag_kmajor(W0, K, lane, k);
    h16 b1 = frag_kmajor(W1, K, lane, k);
    c00 = wmma16x16x32(a0, b0, c00);
    c01 = wmma16x16x32(a0, b1, c01);
    c10 = wmma16x16x32(a1, b0, c10);
    c11 = wmma16x16x32(a1, b1, c11);
  }
  int mlo = (lane & 16) ? 8 : 0;
  int nlo = lane & 15;
#pragma unroll
  for (int tm = 0; tm < 2; ++tm) {
#pragma unroll
    for (int tn = 0; tn < 2; ++tn) {
      f8 cc = tm ? (tn ? c11 : c10) : (tn ? c01 : c00);
      int m0 = (mt + tm) * 16 + mlo;
      int n  = (nt + tn) * 16 + nlo;
      float bn = bias[n];
#pragma unroll
      for (int v = 0; v < 8; ++v) {
        size_t idx = (size_t)(m0 + v) * N + n;
        float val = cc[v] + bn;
        if (mode == 1) {                 // exact GELU: 0.5x(1+erf(x/sqrt(2)))
          val = 0.5f * val * (1.f + erff(val * 0.70710678118654752f));
          out_h[idx] = (_Float16)val;
        } else if (mode == 2) {
          out_f[idx] = resid[idx] + val;
        } else {
          out_h[idx] = (_Float16)val;
        }
      }
    }
  }
}

// ---------------- attention: scores(WMMA) -> masked softmax(LDS) -> P@V(WMMA) ----------------
#define SW_ 1040   // fp16 row stride in LDS (2080B, 16B aligned, bank-spread)
__global__ void attn_kernel(const _Float16* __restrict__ qkv, _Float16* __restrict__ O) {
  int qt = blockIdx.x, h = blockIdx.y, b = blockIdx.z;
  int lane = threadIdx.x & 31, wave = threadIdx.x >> 5;
  const int ld = QKVD_;
  const _Float16* Qbase = qkv + (size_t)b * S_ * ld + h * DH_;
  const _Float16* Kbase = Qbase + D_;
  const _Float16* Vbase = Qbase + 2 * D_;
  int kmax = (qt + 1) * 16;

  __shared__ _Float16 sc[16 * SW_];

  // phase 1: masked, scaled score tiles (causal tiles only)
  const _Float16* Qt = Qbase + (size_t)qt * 16 * ld;
  for (int kt = wave; kt <= qt; kt += 4) {
    const _Float16* Kt = Kbase + (size_t)kt * 16 * ld;
    f8 c = {0.f, 0.f, 0.f, 0.f, 0.f, 0.f, 0.f, 0.f};
    for (int kk = 0; kk < DH_; kk += 32) {
      h16 a  = frag_kmajor(Qt, ld, lane, kk);
      h16 bm = frag_kmajor(Kt, ld, lane, kk);
      c = wmma16x16x32(a, bm, c);
    }
    int mhi = (lane & 16) ? 8 : 0;
    int n   = lane & 15;
    int k_idx = kt * 16 + n;
#pragma unroll
    for (int v = 0; v < 8; ++v) {
      int q_idx = qt * 16 + mhi + v;
      bool causal  = (k_idx <= q_idx);
      bool blocked = ((k_idx >> 2) < (q_idx >> 2)) && ((q_idx & 3) >= 1) &&
                     (((k_idx & 3) == 1) || ((k_idx & 3) == 2));
      float sv = (causal && !blocked) ? c[v] * 0.125f : -1e30f;
      sc[(mhi + v) * SW_ + k_idx] = (_Float16)sv;
    }
  }
  __syncthreads();

  // phase 2: exact softmax per row; 8 lanes per row, shfl_xor reductions (wave32)
  {
    int r = threadIdx.x >> 3, t8 = threadIdx.x & 7;
    _Float16* row = sc + r * SW_;
    float mx = -1e30f;
    for (int ci = t8; ci < kmax; ci += 8) mx = fmaxf(mx, (float)row[ci]);
    for (int msk = 1; msk < 8; msk <<= 1) mx = fmaxf(mx, __shfl_xor(mx, msk, 32));
    float sum = 0.f;
    for (int ci = t8; ci < kmax; ci += 8) {
      float e = __expf((float)row[ci] - mx);
      sum += e;
      row[ci] = (_Float16)e;
    }
    for (int msk = 1; msk < 8; msk <<= 1) sum += __shfl_xor(sum, msk, 32);
    float inv = 1.f / sum;
    for (int ci = t8; ci < kmax; ci += 8) row[ci] = (_Float16)((float)row[ci] * inv);
  }
  // zero-pad P up to a multiple of 32 columns for the WMMA K loop
  int kpad = (kmax + 31) & ~31;
  if (kpad > kmax) {
    for (int i = threadIdx.x; i < 256; i += 128)
      sc[(i >> 4) * SW_ + kmax + (i & 15)] = (_Float16)0.f;
  }
  __syncthreads();

  // phase 3: O = P @ V ; each wave owns a 16-wide slice of DH
  {
    int dcol = wave * 16;
    f8 c = {0.f, 0.f, 0.f, 0.f, 0.f, 0.f, 0.f, 0.f};
    const _Float16* vcol = Vbase + dcol + (lane & 15);
    for (int kc = 0; kc < kpad; kc += 32) {
      h16 a = frag_kmajor(sc, SW_, lane, kc);   // P from LDS in A layout (ds_load_b128)
      int kb2 = kc + ((lane & 16) ? 8 : 0);
      h16 bm;
#pragma unroll
      for (int i = 0; i < 8; ++i) {             // strided V columns
        bm[i]     = vcol[(size_t)(kb2 + i) * ld];
        bm[i + 8] = vcol[(size_t)(kb2 + 16 + i) * ld];
      }
      c = wmma16x16x32(a, bm, c);
    }
    int mhi = (lane & 16) ? 8 : 0;
    int n   = lane & 15;
#pragma unroll
    for (int v = 0; v < 8; ++v) {
      int q_idx = qt * 16 + mhi + v;
      O[((size_t)b * S_ + q_idx) * D_ + h * DH_ + dcol + n] = (_Float16)c[v];
    }
  }
}

// ---------------- host orchestration ----------------
extern "C" void kernel_launch(void* const* d_in, const int* in_sizes, int n_in,
                              void* d_out, int out_size, void* d_ws, size_t ws_size,
                              hipStream_t stream) {
  (void)in_sizes; (void)n_in; (void)out_size; (void)ws_size;
  const int*   tokens  = (const int*)d_in[0];
  const int*   actions = (const int*)d_in[1];
  const float* tok_emb = (const float*)d_in[2];
  const float* act_emb = (const float*)d_in[3];
  const float* lvl_emb = (const float*)d_in[4];
  const float* pos_emb = (const float*)d_in[5];
  const float* ln1_g = (const float*)d_in[6];
  const float* ln1_b = (const float*)d_in[7];
  const float* qkv_w = (const float*)d_in[8];
  const float* qkv_b = (const float*)d_in[9];
  const float* out_w = (const float*)d_in[10];
  const float* out_b = (const float*)d_in[11];
  const float* ln2_g = (const float*)d_in[12];
  const float* ln2_b = (const float*)d_in[13];
  const float* w1    = (const float*)d_in[14];
  const float* b1    = (const float*)d_in[15];
  const float* w2    = (const float*)d_in[16];
  const float* b2    = (const float*)d_in[17];
  const float* lnf_g = (const float*)d_in[18];
  const float* lnf_b = (const float*)d_in[19];

  const size_t M = (size_t)B_ * S_;   // 8192 rows
  char* base = (char*)d_ws;
  size_t off = 0;
  auto carve = [&](size_t bytes) -> void* {
    void* p = base + off;
    off = (off + bytes + 255) & ~(size_t)255;
    return p;
  };
  float*    x      = (float*)   carve(M * D_ * 4);            // fp32 residual stream
  _Float16* xn     = (_Float16*)carve(M * D_ * 2);            // LN output (fp16)
  _Float16* qkvb   = (_Float16*)carve(M * QKVD_ * 2);         // QKV (fp16)
  _Float16* Ob     = (_Float16*)carve(M * D_ * 2);            // attention output (fp16)
  _Float16* hb     = (_Float16*)carve(M * DFF_ * 2);          // FFN hidden (fp16)
  _Float16* qkvw_h = (_Float16*)carve((size_t)L_ * QKVD_ * D_ * 2);
  _Float16* outw_h = (_Float16*)carve((size_t)L_ * D_ * D_ * 2);
  _Float16* w1_h   = (_Float16*)carve((size_t)L_ * DFF_ * D_ * 2);
  _Float16* w2_h   = (_Float16*)carve((size_t)L_ * D_ * DFF_ * 2);

  // fp32 -> fp16 weights (every call; deterministic)
  cvt_f32_f16<<<4096, 256, 0, stream>>>(qkv_w, qkvw_h, (long)L_ * QKVD_ * D_);
  cvt_f32_f16<<<4096, 256, 0, stream>>>(out_w, outw_h, (long)L_ * D_ * D_);
  cvt_f32_f16<<<4096, 256, 0, stream>>>(w1,    w1_h,   (long)L_ * DFF_ * D_);
  cvt_f32_f16<<<4096, 256, 0, stream>>>(w2,    w2_h,   (long)L_ * D_ * DFF_);

  embed_kernel<<<dim3(B_ * S_), 256, 0, stream>>>(tokens, actions, tok_emb, act_emb,
                                                  lvl_emb, pos_emb, x);

  for (int l = 0; l < L_; ++l) {
    ln_kernel<<<(int)M, 256, 0, stream>>>(x, ln1_g + (size_t)l * D_, ln1_b + (size_t)l * D_,
                                          xn, nullptr, 0);
    // QKV: [8192,768] x [2304,768]^T  (64x64 per block)
    gemm_kernel<<<dim3(QKVD_ / 64, (int)M / 64), 128, 0, stream>>>(
        xn, qkvw_h + (size_t)l * QKVD_ * D_, qkv_b + (size_t)l * QKVD_,
        qkvb, nullptr, nullptr, (int)M, QKVD_, D_, 0);
    // attention
    attn_kernel<<<dim3(S_ / 16, H_, B_), 128, 0, stream>>>(qkvb, Ob);
    // out-proj + residual
    gemm_kernel<<<dim3(D_ / 64, (int)M / 64), 128, 0, stream>>>(
        Ob, outw_h + (size_t)l * D_ * D_, out_b + (size_t)l * D_,
        nullptr, x, x, (int)M, D_, D_, 2);
    ln_kernel<<<(int)M, 256, 0, stream>>>(x, ln2_g + (size_t)l * D_, ln2_b + (size_t)l * D_,
                                          xn, nullptr, 0);
    // FFN1 + GELU
    gemm_kernel<<<dim3(DFF_ / 64, (int)M / 64), 128, 0, stream>>>(
        xn, w1_h + (size_t)l * DFF_ * D_, b1 + (size_t)l * DFF_,
        hb, nullptr, nullptr, (int)M, DFF_, D_, 1);
    // FFN2 + residual
    gemm_kernel<<<dim3(D_ / 64, (int)M / 64), 128, 0, stream>>>(
        hb, w2_h + (size_t)l * D_ * DFF_, b2 + (size_t)l * D_,
        nullptr, x, x, (int)M, D_, DFF_, 2);
  }
  // final LN -> fp32 output
  ln_kernel<<<(int)M, 256, 0, stream>>>(x, lnf_g, lnf_b, nullptr, (float*)d_out, 1);
}
